// S4_69148973466319
// MI455X (gfx1250) — matrix operand order repaired and checked by reading, not compiled, not run on hardware
//
#include <hip/hip_runtime.h>

// S4 / SSM layer for MI455X (gfx1250), chunked-scan GEMM formulation.
// B=8, L=8192, D=128, N=128, chunk Q=128, 64 chunks.
// All heavy math via v_wmma_f32_16x16x32_bf16 with bf16x3 (hi/lo) splitting
// for near-fp32 accuracy. Inter-chunk state recurrence is a tiny f32 scan.

#define B_SZ   8
#define L_SEQ  8192
#define D_CH   128
#define N_HID  128
#define Q_CH   128
#define NCHUNK 64
#define COLS   512     // B_SZ * NCHUNK
#define BL     65536   // B_SZ * L_SEQ

typedef __attribute__((ext_vector_type(16))) __bf16 v16bf;
typedef __attribute__((ext_vector_type(8)))  float  v8f;

struct __align__(16) bfx8 { __bf16 v[8]; };

__device__ __forceinline__ float softplus_f(float x) {
  return (x > 20.f) ? x : log1pf(__expf(x));
}

__device__ __forceinline__ void split_bf(float v, __bf16* hi, __bf16* lo) {
  __bf16 h = (__bf16)v;
  *hi = h;
  *lo = (__bf16)(v - (float)h);
}

// ---------------- fragment loaders (per ISA 7.12.2 VGPR layouts) ------------
// A (16x32 bf16, row-major, lda = K stride): lane m = lane&15,
//   e0..7 -> K = kofs..kofs+7, e8..15 -> K = kofs+16..kofs+23, kofs = (lane>=16)?8:0
__device__ __forceinline__ v16bf frag_a(const __bf16* __restrict__ A, int lda,
                                        int k0, int lane) {
  const __bf16* p = A + (size_t)(lane & 15) * lda + k0 + (((lane >> 4) & 1) << 3);
  bfx8 c0 = *(const bfx8*)p;
  bfx8 c1 = *(const bfx8*)(p + 16);
  v16bf r;
#pragma unroll
  for (int i = 0; i < 8; ++i) { r[i] = c0.v[i]; r[i + 8] = c1.v[i]; }
  return r;
}

// B (32x16 bf16, column-major: column stride ldc, K contiguous within column):
//   lane n = lane&15, e0..7 -> K = kofs..kofs+7, e8..15 -> K = kofs+16..kofs+23, kofs = (lane>=16)?8:0
__device__ __forceinline__ v16bf frag_b(const __bf16* __restrict__ Bm, int ldc,
                                        int k0, int lane) {
  const __bf16* p = Bm + (size_t)(lane & 15) * ldc + k0 + (((lane >> 4) & 1) << 3);
  bfx8 c0 = *(const bfx8*)p;
  bfx8 c1 = *(const bfx8*)(p + 16);
  v16bf r;
#pragma unroll
  for (int i = 0; i < 8; ++i) { r[i] = c0.v[i]; r[i + 8] = c1.v[i]; }
  return r;
}

// bf16x3: fp32-ish product from hi/lo split operands.
__device__ __forceinline__ v8f mma3(v16bf ah, v16bf al, v16bf bh, v16bf bl, v8f acc) {
  acc = __builtin_amdgcn_wmma_f32_16x16x32_bf16(false, ah, false, bh, (short)0, acc, false, false);
  acc = __builtin_amdgcn_wmma_f32_16x16x32_bf16(false, ah, false, bl, (short)0, acc, false, false);
  acc = __builtin_amdgcn_wmma_f32_16x16x32_bf16(false, al, false, bh, (short)0, acc, false, false);
  return acc;
}

__device__ __forceinline__ void store_acc(float* p, v8f a) {
  *(float4*)p       = make_float4(a[0], a[1], a[2], a[3]);
  *(float4*)(p + 4) = make_float4(a[4], a[5], a[6], a[7]);
}

// ---------------- K1: per-(d,n) discretization params + W_out split --------
__global__ void s4_setup(const float* __restrict__ log_A, const float* __restrict__ Bmat,
                         const float* __restrict__ Cmat, const float* __restrict__ log_delta,
                         const float* __restrict__ W_out,
                         float* __restrict__ Adt, float* __restrict__ dAQ,
                         float* __restrict__ dBo, float* __restrict__ CdB,
                         __bf16* __restrict__ W2hi, __bf16* __restrict__ W2lo) {
  int d = blockIdx.x, n = threadIdx.x;
  int idx = d * N_HID + n;
  float a   = -softplus_f(log_A[idx]);
  float dt  = softplus_f(log_delta[d]) + 1e-3f;
  float adt = a * dt;                 // log(dA); dA^p = exp(adt*p) exactly
  float dA  = __expf(adt);
  float db  = (dA - 1.0f) / (a + 1e-6f) * Bmat[idx];
  Adt[idx] = adt;
  dAQ[idx] = __expf(adt * (float)Q_CH);
  dBo[idx] = db;
  CdB[idx] = Cmat[idx] * db;
  split_bf(W_out[idx], &W2hi[idx], &W2lo[idx]);
}

// ---------------- K2: per-channel matrices M1, [M2 | T] --------------------
// M1[n,i] = dB_n * dA_n^(Q-1-i)                 (state-input,  N x Q)
// M2[j,n] = C_n  * dA_n^(j+1)                   (state-output, Q x N)
// T[j,i]  = (j>=i) ? k[j-i] : 0,  k[m] = sum_n C_n dB_n dA_n^m   (Q x Q)
__global__ void s4_build(const float* __restrict__ Adt, const float* __restrict__ dBo,
                         const float* __restrict__ CdB, const float* __restrict__ Cmat,
                         __bf16* __restrict__ M1hi, __bf16* __restrict__ M1lo,
                         __bf16* __restrict__ Wdhi, __bf16* __restrict__ Wdlo) {
  __shared__ float sAdt[N_HID], sdB[N_HID], sCdB[N_HID], sC[N_HID], sk[Q_CH];
  int d = blockIdx.x, t = threadIdx.x;
  sAdt[t] = Adt[d * N_HID + t];
  sdB[t]  = dBo[d * N_HID + t];
  sCdB[t] = CdB[d * N_HID + t];
  sC[t]   = Cmat[d * N_HID + t];
  __syncthreads();
  float s = 0.f;
  for (int n = 0; n < N_HID; ++n) s += sCdB[n] * __expf(sAdt[n] * (float)t);
  sk[t] = s;
  __syncthreads();
  // M1 row n=t (row-major, K=Q contiguous)
  {
    size_t base = ((size_t)d * N_HID + t) * Q_CH;
    for (int i = 0; i < Q_CH; ++i) {
      float v = sdB[t] * __expf(sAdt[t] * (float)(Q_CH - 1 - i));
      split_bf(v, &M1hi[base + i], &M1lo[base + i]);
    }
  }
  // Wd row j=t: cols [0,128)=M2, [128,256)=T  (row-major, K=256 contiguous)
  {
    size_t base = ((size_t)d * Q_CH + t) * 256;
    for (int n = 0; n < N_HID; ++n) {
      float v = sC[n] * __expf(sAdt[n] * (float)(t + 1));
      split_bf(v, &Wdhi[base + n], &Wdlo[base + n]);
    }
    for (int i = 0; i < Q_CH; ++i) {
      float v = (t >= i) ? sk[t - i] : 0.f;
      split_bf(v, &Wdhi[base + 128 + i], &Wdlo[base + 128 + i]);
    }
  }
}

// ---------------- K3: x (B,L,D) -> xT hi/lo (D,B,L) via LDS tile -----------
__global__ void s4_transpose_x(const float* __restrict__ x,
                               __bf16* __restrict__ xThi, __bf16* __restrict__ xTlo) {
  __shared__ float t[32][33];
  int l0 = blockIdx.x * 32, d0 = blockIdx.y * 32, b = blockIdx.z;
  int tx = threadIdx.x, ty = threadIdx.y;
#pragma unroll
  for (int it = 0; it < 4; ++it) {
    int lr = ty + it * 8;
    t[lr][tx] = x[((size_t)b * L_SEQ + l0 + lr) * D_CH + d0 + tx];
  }
  __syncthreads();
#pragma unroll
  for (int it = 0; it < 4; ++it) {
    int dr = ty + it * 8;
    float v = t[tx][dr];
    size_t dst = ((size_t)(d0 + dr) * B_SZ + b) * L_SEQ + l0 + tx;
    split_bf(v, &xThi[dst], &xTlo[dst]);
  }
}

// ---------------- K4: U_d (128 x 512) = M1_d @ X_d  (WMMA bf16x3) ----------
// wave -> d = w>>6, m-tile = (w&63)>>3, col-group = (w&7)*64
__global__ void s4_gemm_u(const __bf16* __restrict__ M1hi, const __bf16* __restrict__ M1lo,
                          const __bf16* __restrict__ Xhi,  const __bf16* __restrict__ Xlo,
                          float* __restrict__ U) {
  int lane = threadIdx.x & 31;
  int wave = (blockIdx.x * blockDim.x + threadIdx.x) >> 5;
  int d  = wave >> 6;
  int r  = wave & 63;
  int m0 = (r >> 3) << 4;
  int c0 = (r & 7) << 6;
  const __bf16* Ah = M1hi + (size_t)d * (N_HID * Q_CH) + (size_t)m0 * Q_CH;
  const __bf16* Al = M1lo + (size_t)d * (N_HID * Q_CH) + (size_t)m0 * Q_CH;
  const __bf16* Bh = Xhi + (size_t)d * BL + (size_t)c0 * Q_CH;
  const __bf16* Bl = Xlo + (size_t)d * BL + (size_t)c0 * Q_CH;
  v8f acc[4] = {};
  for (int kk = 0; kk < Q_CH; kk += 32) {
    v16bf ah = frag_a(Ah, Q_CH, kk, lane);
    v16bf al = frag_a(Al, Q_CH, kk, lane);
#pragma unroll
    for (int t = 0; t < 4; ++t) {
      v16bf bh = frag_b(Bh + (size_t)(t * 16) * Q_CH, Q_CH, kk, lane);
      v16bf bl = frag_b(Bl + (size_t)(t * 16) * Q_CH, Q_CH, kk, lane);
      acc[t] = mma3(ah, al, bh, bl, acc[t]);
    }
  }
  int n0 = m0 + (((lane >> 4) & 1) << 3);
#pragma unroll
  for (int t = 0; t < 4; ++t) {
    int col = c0 + t * 16 + (lane & 15);
    store_acc(U + ((size_t)d * COLS + col) * N_HID + n0, acc[t]);
  }
}

// ---------------- K5: inter-chunk state scan (f32) -------------------------
// H_0 = 0 ; H_{c} written ; h = dA^Q * h + U_c.  Layout (D, COLS, N), n fastest.
__global__ void s4_scan(const float* __restrict__ U, const float* __restrict__ dAQ,
                        __bf16* __restrict__ Hhi, __bf16* __restrict__ Hlo) {
  int tid = blockIdx.x * blockDim.x + threadIdx.x;   // D*B*N = 131072
  int n = tid & 127;
  int b = (tid >> 7) & 7;
  int d = tid >> 10;
  float a = dAQ[d * N_HID + n];
  float h = 0.f;
  size_t base = ((size_t)d * COLS + (size_t)b * NCHUNK) * N_HID + n;
  for (int c = 0; c < NCHUNK; ++c) {
    size_t idx = base + (size_t)c * N_HID;
    __bf16 hi = (__bf16)h;
    Hhi[idx] = hi;
    Hlo[idx] = (__bf16)(h - (float)hi);
    h = a * h + U[idx];
  }
}

// ---------------- K6: Y_d = [M2|T] @ [H;X], fused skip -> Zf (D,B,L) f32 ---
__global__ void s4_gemm_y(const __bf16* __restrict__ Wdhi, const __bf16* __restrict__ Wdlo,
                          const __bf16* __restrict__ Hhi,  const __bf16* __restrict__ Hlo,
                          const __bf16* __restrict__ Xhi,  const __bf16* __restrict__ Xlo,
                          const float* __restrict__ skip_D, float* __restrict__ Zf) {
  int lane = threadIdx.x & 31;
  int wave = (blockIdx.x * blockDim.x + threadIdx.x) >> 5;
  int d  = wave >> 6;
  int r  = wave & 63;
  int m0 = (r >> 3) << 4;
  int c0 = (r & 7) << 6;
  const __bf16* Ah = Wdhi + (size_t)d * (Q_CH * 256) + (size_t)m0 * 256;
  const __bf16* Al = Wdlo + (size_t)d * (Q_CH * 256) + (size_t)m0 * 256;
  v8f acc[4] = {};
  for (int kk = 0; kk < 256; kk += 32) {
    v16bf ah = frag_a(Ah, 256, kk, lane);
    v16bf al = frag_a(Al, 256, kk, lane);
    const __bf16 *bhb, *blb;
    int kl;
    if (kk < 128) {              // state rows: H (D, COLS, N), col stride 128
      bhb = Hhi + (size_t)d * ((size_t)COLS * N_HID);
      blb = Hlo + (size_t)d * ((size_t)COLS * N_HID);
      kl = kk;
    } else {                     // input rows: xT (D,B,L), col stride 128
      bhb = Xhi + (size_t)d * BL;
      blb = Xlo + (size_t)d * BL;
      kl = kk - 128;
    }
#pragma unroll
    for (int t = 0; t < 4; ++t) {
      v16bf bh = frag_b(bhb + (size_t)(c0 + t * 16) * 128, 128, kl, lane);
      v16bf bl = frag_b(blb + (size_t)(c0 + t * 16) * 128, 128, kl, lane);
      acc[t] = mma3(ah, al, bh, bl, acc[t]);
    }
  }
  float sk = skip_D[d];
  int j0 = m0 + (((lane >> 4) & 1) << 3);
#pragma unroll
  for (int t = 0; t < 4; ++t) {
    int col = c0 + t * 16 + (lane & 15);
    size_t a0 = (size_t)d * BL + (size_t)col * 128 + j0;
    bfx8 xh = *(const bfx8*)(Xhi + a0);
    bfx8 xl = *(const bfx8*)(Xlo + a0);
    v8f z = acc[t];
#pragma unroll
    for (int i = 0; i < 8; ++i) z[i] += sk * ((float)xh.v[i] + (float)xl.v[i]);
    store_acc(Zf + a0, z);
  }
}

// ---------------- K7: Zf (D,B,L) f32 -> Zt hi/lo (B,L,D) -------------------
__global__ void s4_transpose_z(const float* __restrict__ Zf,
                               __bf16* __restrict__ Zthi, __bf16* __restrict__ Ztlo) {
  __shared__ float t[32][33];
  int l0 = blockIdx.x * 32, d0 = blockIdx.y * 32, b = blockIdx.z;
  int tx = threadIdx.x, ty = threadIdx.y;
#pragma unroll
  for (int it = 0; it < 4; ++it) {
    int dr = ty + it * 8;
    t[dr][tx] = Zf[((size_t)(d0 + dr) * B_SZ + b) * L_SEQ + l0 + tx];
  }
  __syncthreads();
#pragma unroll
  for (int it = 0; it < 4; ++it) {
    int lr = ty + it * 8;
    float v = t[tx][lr];
    size_t dst = ((size_t)b * L_SEQ + l0 + lr) * D_CH + d0 + tx;
    split_bf(v, &Zthi[dst], &Ztlo[dst]);
  }
}

// ---------------- K8: out(bl,d') = W_out @ z + b_out  (WMMA bf16x3) --------
__global__ void s4_gemm_out(const __bf16* __restrict__ W2hi, const __bf16* __restrict__ W2lo,
                            const __bf16* __restrict__ Zthi, const __bf16* __restrict__ Ztlo,
                            const float* __restrict__ b_out, float* __restrict__ out) {
  int lane = threadIdx.x & 31;
  int wave = (blockIdx.x * blockDim.x + threadIdx.x) >> 5;
  int mt = wave >> 10;        // 8 m-tiles
  int cg = wave & 1023;       // 1024 col groups of 64
  int m0 = mt << 4;
  int c0 = cg << 6;
  const __bf16* Ah = W2hi + (size_t)m0 * D_CH;
  const __bf16* Al = W2lo + (size_t)m0 * D_CH;
  v8f acc[4] = {};
  for (int kk = 0; kk < D_CH; kk += 32) {
    v16bf ah = frag_a(Ah, D_CH, kk, lane);
    v16bf al = frag_a(Al, D_CH, kk, lane);
#pragma unroll
    for (int t = 0; t < 4; ++t) {
      v16bf bh = frag_b(Zthi + (size_t)(c0 + t * 16) * D_CH, D_CH, kk, lane);
      v16bf bl = frag_b(Ztlo + (size_t)(c0 + t * 16) * D_CH, D_CH, kk, lane);
      acc[t] = mma3(ah, al, bh, bl, acc[t]);
    }
  }
  int dd = m0 + (((lane >> 4) & 1) << 3);
  float4 b0 = *(const float4*)(b_out + dd);
  float4 b1 = *(const float4*)(b_out + dd + 4);
#pragma unroll
  for (int t = 0; t < 4; ++t) {
    int bl = c0 + t * 16 + (lane & 15);
    float* p = out + (size_t)bl * D_CH + dd;
    *(float4*)p       = make_float4(acc[t][0] + b0.x, acc[t][1] + b0.y,
                                    acc[t][2] + b0.z, acc[t][3] + b0.w);
    *(float4*)(p + 4) = make_float4(acc[t][4] + b1.x, acc[t][5] + b1.y,
                                    acc[t][6] + b1.z, acc[t][7] + b1.w);
  }
}

// ---------------------------------------------------------------------------
extern "C" void kernel_launch(void* const* d_in, const int* in_sizes, int n_in,
                              void* d_out, int out_size, void* d_ws, size_t ws_size,
                              hipStream_t stream) {
  (void)in_sizes; (void)n_in; (void)out_size; (void)ws_size;
  const float* x         = (const float*)d_in[0];
  const float* log_A     = (const float*)d_in[1];
  const float* Bmat      = (const float*)d_in[2];
  const float* Cmat      = (const float*)d_in[3];
  const float* log_delta = (const float*)d_in[4];
  const float* skip_D    = (const float*)d_in[5];
  const float* W_out     = (const float*)d_in[6];
  const float* b_out     = (const float*)d_in[7];
  float* out = (float*)d_out;

  char* ws = (char*)d_ws;
  size_t o = 0;
  auto al = [&](size_t bytes) { size_t r = o; o += (bytes + 255) & ~(size_t)255; return r; };
  float*  Adt  = (float*)(ws + al(65536));
  float*  dAQ  = (float*)(ws + al(65536));
  float*  dBo  = (float*)(ws + al(65536));
  float*  CdB  = (float*)(ws + al(65536));
  __bf16* W2hi = (__bf16*)(ws + al(32768));
  __bf16* W2lo = (__bf16*)(ws + al(32768));
  __bf16* M1hi = (__bf16*)(ws + al(4194304));
  __bf16* M1lo = (__bf16*)(ws + al(4194304));
  __bf16* Wdhi = (__bf16*)(ws + al(8388608));
  __bf16* Wdlo = (__bf16*)(ws + al(8388608));
  __bf16* xThi = (__bf16*)(ws + al(16777216));
  __bf16* xTlo = (__bf16*)(ws + al(16777216));
  float*  U    = (float*)(ws + al(33554432));   // reused as Zf after the scan
  __bf16* Hhi  = (__bf16*)(ws + al(16777216));
  __bf16* Hlo  = (__bf16*)(ws + al(16777216));
  __bf16* Zthi = (__bf16*)(ws + al(16777216));
  __bf16* Ztlo = (__bf16*)(ws + al(16777216));
  float*  Zf   = U;  // U is dead after s4_scan; alias to save scratch

  s4_setup<<<dim3(D_CH), dim3(N_HID), 0, stream>>>(log_A, Bmat, Cmat, log_delta, W_out,
                                                   Adt, dAQ, dBo, CdB, W2hi, W2lo);
  s4_build<<<dim3(D_CH), dim3(Q_CH), 0, stream>>>(Adt, dBo, CdB, Cmat,
                                                  M1hi, M1lo, Wdhi, Wdlo);
  s4_transpose_x<<<dim3(L_SEQ / 32, D_CH / 32, B_SZ), dim3(32, 8), 0, stream>>>(x, xThi, xTlo);
  s4_gemm_u<<<dim3(1024), dim3(256), 0, stream>>>(M1hi, M1lo, xThi, xTlo, U);
  s4_scan<<<dim3(512), dim3(256), 0, stream>>>(U, dAQ, Hhi, Hlo);
  s4_gemm_y<<<dim3(1024), dim3(256), 0, stream>>>(Wdhi, Wdlo, Hhi, Hlo, xThi, xTlo,
                                                  skip_D, Zf);
  s4_transpose_z<<<dim3(L_SEQ / 32, D_CH / 32, B_SZ), dim3(32, 8), 0, stream>>>(Zf, Zthi, Ztlo);
  s4_gemm_out<<<dim3(1024), dim3(256), 0, stream>>>(W2hi, W2lo, Zthi, Ztlo, b_out, out);
}
